// GCN_MCAA_11974368821654
// MI455X (gfx1250) — matrix operand order repaired and checked by reading, not compiled, not run on hardware
//
#include <hip/hip_runtime.h>
#include <math.h>

#define GN 2048
#define GC 3
#define GE 4
#define GCELL 15

typedef __attribute__((ext_vector_type(16))) _Float16 v16h;
typedef __attribute__((ext_vector_type(8)))  _Float16 v8h;
typedef __attribute__((ext_vector_type(8)))  float    v8f;

// ---------------------------------------------------------------------------
// Generic small matmul: out[n,k] = sum_f in[n,f] * W[f,k]   (rows = GN)
// ---------------------------------------------------------------------------
__global__ __launch_bounds__(256) void k_proj(const float* __restrict__ X, int F,
                                              const float* __restrict__ W, int K,
                                              float* __restrict__ out)
{
  int idx = blockIdx.x * 256 + threadIdx.x;
  if (idx >= GN * K) return;
  int n = idx / K, k = idx % K;
  const float* xr = X + (size_t)n * F;
  float s = 0.f;
  for (int f = 0; f < F; ++f) s += xr[f] * W[f * K + k];
  out[(size_t)n * K + k] = s;
}

// ---------------------------------------------------------------------------
// h[n,k] = tanh( sum_m R[n,m] * W[m,k] + 0.1 ),  k < 15
// ---------------------------------------------------------------------------
__global__ __launch_bounds__(256) void k_dense_tanh(const float* __restrict__ R, int M,
                                                    const float* __restrict__ W,
                                                    float* __restrict__ h)
{
  int idx = blockIdx.x * 256 + threadIdx.x;
  if (idx >= GN * GCELL) return;
  int n = idx / GCELL, k = idx % GCELL;
  const float* r = R + (size_t)n * M;
  float s = 0.1f;
  for (int m = 0; m < M; ++m) s += r[m] * W[m * GCELL + k];
  h[idx] = tanhf(s);
}

// ---------------------------------------------------------------------------
// Column-block copy: dst[n, off+c] = src[n, c]
// ---------------------------------------------------------------------------
__global__ __launch_bounds__(256) void k_copy_cols(const float* __restrict__ src, int lds,
                                                   float* __restrict__ dst, int ldd,
                                                   int off, int ncols)
{
  int idx = blockIdx.x * 256 + threadIdx.x;
  if (idx >= GN * ncols) return;
  int n = idx / ncols, c = idx % ncols;
  dst[(size_t)n * ldd + off + c] = src[(size_t)n * lds + c];
}

// ---------------------------------------------------------------------------
// Transpose f32 [GN, F] -> f16 [FP, GN], zero-padding rows F..FP-1
// ---------------------------------------------------------------------------
__global__ __launch_bounds__(256) void k_t16(const float* __restrict__ src, int F,
                                             _Float16* __restrict__ dst, int FP)
{
  int idx = blockIdx.x * 256 + threadIdx.x;
  if (idx >= FP * GN) return;
  int f = idx / GN, n = idx % GN;
  dst[idx] = (f < F) ? (_Float16)src[(size_t)n * F + f] : (_Float16)0.f;
}

// ---------------------------------------------------------------------------
// Pairwise kernel: one workgroup per row i.
//  pass1: masked att scores v -> LDS, local max; gate branch accumulation
//  pass2: exact softmax denominator from LDS
//  pass3: att f16 out (GEMM A layout [c][i][j]), y_edge accumulation
// Writes y_gate -> R[i, gate_off..], y_edge -> R[i, edge_off..]
// ---------------------------------------------------------------------------
__global__ __launch_bounds__(256) void k_pairwise(
    const float* __restrict__ A, const float* __restrict__ Attr,
    const float* __restrict__ xa0, const float* __restrict__ xa1,
    const float* __restrict__ ae,
    const float* __restrict__ xi0, const float* __restrict__ xi1,
    const float* __restrict__ ie,
    const float* __restrict__ xg0, const float* __restrict__ xg1,
    const float* __restrict__ ge,
    _Float16* __restrict__ attout,
    float* __restrict__ R, int ldR, int edge_off, int gate_off)
{
  __shared__ float sV[GN * GC];          // 24 KB masked scores for this row
  __shared__ float red[256 * 16];        // 16 KB reduction scratch
  __shared__ float sw[GE * GC + 2 * GE * GCELL];   // ae(12) | ie(60) | ge(60)
  __shared__ float srow[GC + 2 * GCELL];           // xa0[i] | xi0[i] | xg0[i]
  __shared__ float sm[GC], sd[GC];

  const int i = blockIdx.x;
  const int t = threadIdx.x;

  if (t < GE * GC)    sw[t]       = ae[t];
  if (t < GE * GCELL) sw[12 + t]  = ie[t];
  if (t < GE * GCELL) sw[72 + t]  = ge[t];
  if (t < GC)         srow[t]                 = xa0[i * GC + t];
  if (t < GCELL)      srow[GC + t]            = xi0[i * GCELL + t];
  if (t < GCELL)      srow[GC + GCELL + t]    = xg0[i * GCELL + t];
  __syncthreads();

  float lmax[GC];
  float gacc[GCELL];
  #pragma unroll
  for (int c = 0; c < GC; ++c) lmax[c] = 0.f;      // masked entries give v = 0
  #pragma unroll
  for (int k = 0; k < GCELL; ++k) gacc[k] = 0.f;

  const float4* attr4 = (const float4*)Attr;

  // ---- pass 1
  for (int j = t; j < GN; j += 256) {
    float  a = A[(size_t)i * GN + j];
    float4 e = attr4[(size_t)i * GN + j];
    #pragma unroll
    for (int c = 0; c < GC; ++c) {
      float s = srow[c] + xa1[j * GC + c]
              + e.x * sw[0 * GC + c] + e.y * sw[1 * GC + c]
              + e.z * sw[2 * GC + c] + e.w * sw[3 * GC + c];
      float v = ((s > 0.f) ? s : (expf(s) - 1.f)) + 1.0001f;   // elu + 1.0001
      v *= a;
      sV[j * GC + c] = v;
      lmax[c] = fmaxf(lmax[c], v);
    }
    #pragma unroll
    for (int k = 0; k < GCELL; ++k) {
      float si = srow[GC + k] + xi1[j * GCELL + k]
               + e.x * sw[12 + 0 * GCELL + k] + e.y * sw[12 + 1 * GCELL + k]
               + e.z * sw[12 + 2 * GCELL + k] + e.w * sw[12 + 3 * GCELL + k];
      float sg = srow[GC + GCELL + k] + xg1[j * GCELL + k]
               + e.x * sw[72 + 0 * GCELL + k] + e.y * sw[72 + 1 * GCELL + k]
               + e.z * sw[72 + 2 * GCELL + k] + e.w * sw[72 + 3 * GCELL + k];
      gacc[k] += tanhf(si) / (1.f + expf(-sg));    // info * sigmoid(gate)
    }
  }

  // ---- reduce max per channel
  #pragma unroll
  for (int c = 0; c < GC; ++c) red[t * 16 + c] = lmax[c];
  __syncthreads();
  for (int s = 128; s > 0; s >>= 1) {
    if (t < s) {
      #pragma unroll
      for (int c = 0; c < GC; ++c)
        red[t * 16 + c] = fmaxf(red[t * 16 + c], red[(t + s) * 16 + c]);
    }
    __syncthreads();
  }
  if (t < GC) sm[t] = red[t];
  __syncthreads();

  // ---- pass 2: softmax denominator (includes masked zeros, as in reference)
  float lsum[GC] = {0.f, 0.f, 0.f};
  for (int j = t; j < GN; j += 256) {
    #pragma unroll
    for (int c = 0; c < GC; ++c) lsum[c] += expf(sV[j * GC + c] - sm[c]);
  }
  #pragma unroll
  for (int c = 0; c < GC; ++c) red[t * 16 + c] = lsum[c];
  __syncthreads();
  for (int s = 128; s > 0; s >>= 1) {
    if (t < s) {
      #pragma unroll
      for (int c = 0; c < GC; ++c) red[t * 16 + c] += red[(t + s) * 16 + c];
    }
    __syncthreads();
  }
  if (t < GC) sd[t] = red[t];
  __syncthreads();

  float inv[GC];
  #pragma unroll
  for (int c = 0; c < GC; ++c) inv[c] = 1.f / sd[c];

  // ---- pass 3: final att (re-masked), att f16 out, y_edge accumulation
  float eacc[GC * GE];
  #pragma unroll
  for (int q = 0; q < GC * GE; ++q) eacc[q] = 0.f;

  for (int j = t; j < GN; j += 256) {
    float  a = A[(size_t)i * GN + j];
    float4 e = attr4[(size_t)i * GN + j];
    #pragma unroll
    for (int c = 0; c < GC; ++c) {
      float att = expf(sV[j * GC + c] - sm[c]) * inv[c] * a;
      attout[(size_t)c * GN * GN + (size_t)i * GN + j] = (_Float16)att;
      eacc[c * GE + 0] += att * e.x;
      eacc[c * GE + 1] += att * e.y;
      eacc[c * GE + 2] += att * e.z;
      eacc[c * GE + 3] += att * e.w;
    }
  }

  // ---- reduce y_gate (15)
  __syncthreads();
  #pragma unroll
  for (int k = 0; k < 16; ++k) red[t * 16 + k] = (k < GCELL) ? gacc[k] : 0.f;
  __syncthreads();
  for (int s = 128; s > 0; s >>= 1) {
    if (t < s) {
      #pragma unroll
      for (int k = 0; k < 16; ++k) red[t * 16 + k] += red[(t + s) * 16 + k];
    }
    __syncthreads();
  }
  if (t < GCELL) R[(size_t)i * ldR + gate_off + t] = red[t];
  __syncthreads();

  // ---- reduce y_edge (12)
  #pragma unroll
  for (int k = 0; k < 16; ++k) red[t * 16 + k] = (k < GC * GE) ? eacc[k] : 0.f;
  __syncthreads();
  for (int s = 128; s > 0; s >>= 1) {
    if (t < s) {
      #pragma unroll
      for (int k = 0; k < 16; ++k) red[t * 16 + k] += red[(t + s) * 16 + k];
    }
    __syncthreads();
  }
  if (t < GC * GE) R[(size_t)i * ldR + edge_off + t] = red[t];
}

// ---------------------------------------------------------------------------
// WMMA GEMM: y_node[c] = att_c [GN x GN] (f16) @ B [GN x Fcols] (B given as
// BT f16 [FP x GN]).  One wave per 16x16 output tile, K-loop of 32.
// Output scattered into R[:, outoff + c*Fcols + n].
// ---------------------------------------------------------------------------
__global__ __launch_bounds__(32) void k_wmma_ynode(
    const _Float16* __restrict__ Amat,   // [GC][GN][GN]
    const _Float16* __restrict__ BT,     // [FP][GN]
    float* __restrict__ R, int ldR, int outoff, int Fcols, int nvalid)
{
  const int lane = threadIdx.x & 31;
  const int m0 = blockIdx.x * 16;
  const int n0 = blockIdx.y * 16;
  const int c  = blockIdx.z;
  const int lm = lane & 15;
  const int hi = lane >> 4;

  const _Float16* arow = Amat + (size_t)c * GN * GN + (size_t)(m0 + lm) * GN + hi * 16;
  const _Float16* bcol = BT + (size_t)(n0 + lm) * GN + hi * 16;

  v8f acc = {};
  for (int k0 = 0; k0 < GN; k0 += 32) {
    // A fragment (16x32 f16, wave32 layout): halves 0..15 -> K = hi*16 + 0..15
    // (contiguous, matching B)
    v8h alo = *(const v8h*)(arow + k0);
    v8h ahi = *(const v8h*)(arow + k0 + 8);
    v16h a;
    #pragma unroll
    for (int q = 0; q < 8; ++q) { a[q] = alo[q]; a[q + 8] = ahi[q]; }
    // B fragment (32x16 f16): lane = (K/16)*16 + n, K-contiguous from BT row
    v16h b = *(const v16h*)(bcol + k0);
    acc = __builtin_amdgcn_wmma_f32_16x16x32_f16(
        false, a, false, b, (short)0, acc, false, false);
  }

  if (n0 + lm < nvalid) {
    #pragma unroll
    for (int r = 0; r < 8; ++r) {
      int row = m0 + hi * 8 + r;                       // f32 C/D layout
      R[(size_t)row * ldR + outoff + c * Fcols + n0 + lm] = acc[r];
    }
  }
}

// ---------------------------------------------------------------------------
// Softmax over node axis (axis=0), one workgroup per column k, in-place.
// ---------------------------------------------------------------------------
__global__ __launch_bounds__(256) void k_softmax_col(float* __restrict__ s, int K)
{
  __shared__ float red[256];
  __shared__ float mm, dd;
  int k = blockIdx.x, t = threadIdx.x;
  float m = -3.4e38f;
  for (int n = t; n < GN; n += 256) m = fmaxf(m, s[(size_t)n * K + k]);
  red[t] = m; __syncthreads();
  for (int st = 128; st > 0; st >>= 1) {
    if (t < st) red[t] = fmaxf(red[t], red[t + st]);
    __syncthreads();
  }
  if (t == 0) mm = red[0];
  __syncthreads();
  float d = 0.f;
  for (int n = t; n < GN; n += 256) d += expf(s[(size_t)n * K + k] - mm);
  red[t] = d; __syncthreads();
  for (int st = 128; st > 0; st >>= 1) {
    if (t < st) red[t] += red[t + st];
    __syncthreads();
  }
  if (t == 0) dd = red[0];
  __syncthreads();
  float invd = 1.f / dd;
  for (int n = t; n < GN; n += 256)
    s[(size_t)n * K + k] = expf(s[(size_t)n * K + k] - mm) * invd;
}

// ---------------------------------------------------------------------------
// image[k,f] = sum_n att[n,k] * Xc[n,f]
// ---------------------------------------------------------------------------
__global__ __launch_bounds__(128) void k_pool(const float* __restrict__ att,
                                              const float* __restrict__ Xc,
                                              float* __restrict__ image)
{
  int k = blockIdx.x, f = threadIdx.x;
  if (f >= 94) return;
  float s = 0.f;
  for (int n = 0; n < GN; ++n) s += att[(size_t)n * 10 + k] * Xc[(size_t)n * 94 + f];
  image[k * 94 + f] = s;
}

// ---------------------------------------------------------------------------
// Head: image2 = tanh(image @ out_cnn + 0.1) [10,15]; logits = flat @ out_r
// + 0.1; softmax(2) -> d_out
// ---------------------------------------------------------------------------
__global__ __launch_bounds__(256) void k_head(const float* __restrict__ image,
                                              const float* __restrict__ out_cnn,
                                              const float* __restrict__ out_r,
                                              float* __restrict__ out)
{
  __shared__ float sim[160];
  int t = threadIdx.x;
  if (t < 150) {
    int k = t / 15, o = t % 15;
    float s = 0.1f;
    for (int f = 0; f < 94; ++f) s += image[k * 94 + f] * out_cnn[f * 15 + o];
    sim[t] = tanhf(s);
  }
  __syncthreads();
  if (t == 0) {
    float l0 = 0.1f, l1 = 0.1f;
    for (int m = 0; m < 150; ++m) { l0 += sim[m] * out_r[m * 2]; l1 += sim[m] * out_r[m * 2 + 1]; }
    float mx = fmaxf(l0, l1);
    float e0 = expf(l0 - mx), e1 = expf(l1 - mx);
    float invd = 1.f / (e0 + e1);
    out[0] = e0 * invd;
    out[1] = e1 * invd;
  }
}

// ---------------------------------------------------------------------------
extern "C" void kernel_launch(void* const* d_in, const int* in_sizes, int n_in,
                              void* d_out, int out_size, void* d_ws, size_t ws_size,
                              hipStream_t stream)
{
  (void)in_sizes; (void)n_in; (void)out_size; (void)ws_size;

  const float* X    = (const float*)d_in[0];
  const float* A    = (const float*)d_in[1];
  const float* Attr = (const float*)d_in[2];

  // workspace carve-out
  char* ws = (char*)d_ws;
  size_t off = 0;
  auto alloc = [&](size_t bytes) -> char* {
    char* p = ws + off;
    off = (off + bytes + 255) & ~(size_t)255;
    return p;
  };
  _Float16* attbuf = (_Float16*)alloc((size_t)GC * GN * GN * 2);  // 24 MB
  _Float16* XT     = (_Float16*)alloc((size_t)64 * GN * 2);
  _Float16* hT     = (_Float16*)alloc((size_t)16 * GN * 2);
  float* R0  = (float*)alloc((size_t)GN * 283 * 4);
  float* R1  = (float*)alloc((size_t)GN * 87 * 4);
  float* h0  = (float*)alloc((size_t)GN * 15 * 4);
  float* h1  = (float*)alloc((size_t)GN * 15 * 4);
  float* xa0 = (float*)alloc((size_t)GN * GC * 4);
  float* xa1 = (float*)alloc((size_t)GN * GC * 4);
  float* xi0 = (float*)alloc((size_t)GN * GCELL * 4);
  float* xi1 = (float*)alloc((size_t)GN * GCELL * 4);
  float* xg0 = (float*)alloc((size_t)GN * GCELL * 4);
  float* xg1 = (float*)alloc((size_t)GN * GCELL * 4);
  float* Xc     = (float*)alloc((size_t)GN * 94 * 4);
  float* scores = (float*)alloc((size_t)GN * 10 * 4);
  float* image  = (float*)alloc((size_t)10 * 94 * 4);

  dim3 b256(256);
  auto g1 = [](int total) { return dim3((unsigned)((total + 255) / 256)); };

  // ======================= Layer 0 =======================
  k_proj<<<g1(GN * GC),    b256, 0, stream>>>(X, 64, (const float*)d_in[3],  GC,    xa0);
  k_proj<<<g1(GN * GC),    b256, 0, stream>>>(X, 64, (const float*)d_in[4],  GC,    xa1);
  k_proj<<<g1(GN * GCELL), b256, 0, stream>>>(X, 64, (const float*)d_in[6],  GCELL, xi0);
  k_proj<<<g1(GN * GCELL), b256, 0, stream>>>(X, 64, (const float*)d_in[7],  GCELL, xi1);
  k_proj<<<g1(GN * GCELL), b256, 0, stream>>>(X, 64, (const float*)d_in[9],  GCELL, xg0);
  k_proj<<<g1(GN * GCELL), b256, 0, stream>>>(X, 64, (const float*)d_in[10], GCELL, xg1);

  // R0 layout: X[0:64) | y_edge[64:76) | y_node[76:268) | y_gate[268:283)
  k_pairwise<<<dim3(GN), b256, 0, stream>>>(
      A, Attr, xa0, xa1, (const float*)d_in[5],
      xi0, xi1, (const float*)d_in[8],
      xg0, xg1, (const float*)d_in[11],
      attbuf, R0, 283, 64, 268);

  k_copy_cols<<<g1(GN * 64), b256, 0, stream>>>(X, 64, R0, 283, 0, 64);
  k_t16<<<g1(64 * GN), b256, 0, stream>>>(X, 64, XT, 64);

  k_wmma_ynode<<<dim3(GN / 16, 4, GC), dim3(32), 0, stream>>>(
      attbuf, XT, R0, 283, 76, 64, 64);

  k_dense_tanh<<<g1(GN * GCELL), b256, 0, stream>>>(R0, 283, (const float*)d_in[12], h0);

  // ======================= Layer 1 =======================
  k_proj<<<g1(GN * GC),    b256, 0, stream>>>(h0, 15, (const float*)d_in[13], GC,    xa0);
  k_proj<<<g1(GN * GC),    b256, 0, stream>>>(h0, 15, (const float*)d_in[14], GC,    xa1);
  k_proj<<<g1(GN * GCELL), b256, 0, stream>>>(h0, 15, (const float*)d_in[16], GCELL, xi0);
  k_proj<<<g1(GN * GCELL), b256, 0, stream>>>(h0, 15, (const float*)d_in[17], GCELL, xi1);
  k_proj<<<g1(GN * GCELL), b256, 0, stream>>>(h0, 15, (const float*)d_in[19], GCELL, xg0);
  k_proj<<<g1(GN * GCELL), b256, 0, stream>>>(h0, 15, (const float*)d_in[20], GCELL, xg1);

  // R1 layout: h0[0:15) | y_edge[15:27) | y_node[27:72) | y_gate[72:87)
  k_pairwise<<<dim3(GN), b256, 0, stream>>>(
      A, Attr, xa0, xa1, (const float*)d_in[15],
      xi0, xi1, (const float*)d_in[18],
      xg0, xg1, (const float*)d_in[21],
      attbuf, R1, 87, 15, 72);

  k_copy_cols<<<g1(GN * 15), b256, 0, stream>>>(h0, 15, R1, 87, 0, 15);
  k_t16<<<g1(16 * GN), b256, 0, stream>>>(h0, 15, hT, 16);

  k_wmma_ynode<<<dim3(GN / 16, 1, GC), dim3(32), 0, stream>>>(
      attbuf, hT, R1, 87, 27, 15, 15);

  k_dense_tanh<<<g1(GN * GCELL), b256, 0, stream>>>(R1, 87, (const float*)d_in[22], h1);

  // ======================= Head =======================
  k_copy_cols<<<g1(GN * 64), b256, 0, stream>>>(X,  64, Xc, 94, 0,  64);
  k_copy_cols<<<g1(GN * 15), b256, 0, stream>>>(h0, 15, Xc, 94, 64, 15);
  k_copy_cols<<<g1(GN * 15), b256, 0, stream>>>(h1, 15, Xc, 94, 79, 15);

  k_proj<<<g1(GN * 10), b256, 0, stream>>>(Xc, 94, (const float*)d_in[23], 10, scores);
  k_softmax_col<<<dim3(10), b256, 0, stream>>>(scores, 10);
  k_pool<<<dim3(10), dim3(128), 0, stream>>>(scores, Xc, image);
  k_head<<<dim3(1), b256, 0, stream>>>(image, (const float*)d_in[24],
                                       (const float*)d_in[25], (float*)d_out);
}
